// TSPGAT_84645215469795
// MI455X (gfx1250) — matrix operand order repaired and checked by reading, not compiled, not run on hardware
//
#include <hip/hip_runtime.h>
#include <hip/hip_bf16.h>
#include <math.h>

typedef __attribute__((ext_vector_type(16))) __bf16 v16bf;
typedef __attribute__((ext_vector_type(8)))  float  v8f;
typedef __attribute__((ext_vector_type(4)))  unsigned int u32x4;

#define HIDC 128
#define NEG_SLOPE 0.2f

// ---------------- small utility kernels ----------------
__global__ void k_cvt_bf16(const float* __restrict__ src, __bf16* __restrict__ dst, int n) {
  int i = blockIdx.x * blockDim.x + threadIdx.x;
  if (i < n) dst[i] = (__bf16)src[i];
}

__global__ void k_fill(float* __restrict__ p, float v, int n) {
  int i = blockIdx.x * blockDim.x + threadIdx.x;
  if (i < n) p[i] = v;
}

// ---------------- WMMA GEMM: out[N,M] = A[N,K](bf16) @ W[M,K]^T (bf16), f32 acc ----------------
// A-fragment (16x32 bf16, ISA interleaved layout): lane L holds row r0+(L&15);
// lanes 0-15: K = k0..k0+7 (v[0..7]) and k0+16..k0+23 (v[8..15]); lanes 16-31: +8.
__device__ __forceinline__ v16bf frag_a(const __bf16* base, int row, int K, int k0, int lane) {
  const __bf16* p = base + (size_t)row * K + k0 + ((lane >> 4) << 3);
  union { v16bf v; u32x4 q[2]; } u;
  u.q[0] = *(const u32x4*)p;
  u.q[1] = *(const u32x4*)(p + 16);
  return u.v;
}
// B-fragment (32x16 bf16): lane L holds col c0+(L&15); lanes 0-15: K = k0..k0+7 (v[0..7])
// and k0+16..k0+23 (v[8..15]); lanes 16-31: +8 (same interleave as A).
__device__ __forceinline__ v16bf frag_b(const __bf16* base, int col, int K, int k0, int lane) {
  const __bf16* p = base + (size_t)col * K + k0 + ((lane >> 4) << 3);
  union { v16bf v; u32x4 q[2]; } u;
  u.q[0] = *(const u32x4*)p;
  u.q[1] = *(const u32x4*)(p + 16);
  return u.v;
}

__global__ void k_gemm_wmma(const __bf16* __restrict__ A, const __bf16* __restrict__ W,
                            float* __restrict__ out, int Nr, int K, int M) {
  const int lane = threadIdx.x & 31;
  const int wave = threadIdx.x >> 5;
  const int Mt = M >> 4;
  const int Nt = (Nr + 15) >> 4;
  const int tile = blockIdx.x * (blockDim.x >> 5) + wave;  // wave-uniform
  if (tile >= Nt * Mt) return;                              // uniform branch: EXEC stays full
  const int tn = tile / Mt;
  const int tm = tile - tn * Mt;
  const int r0 = tn << 4, c0 = tm << 4;
  const int r16 = lane & 15;
  int arow = r0 + r16; if (arow >= Nr) arow = Nr - 1;        // clamp loads; stores guarded
  const int bcol = c0 + r16;
  v8f acc = {};
  for (int k0 = 0; k0 < K; k0 += 32) {
    v16bf a = frag_a(A, arow, K, k0, lane);
    v16bf b = frag_b(W, bcol, K, k0, lane);
    acc = __builtin_amdgcn_wmma_f32_16x16x32_bf16(false, a, false, b, (short)0, acc,
                                                  false, false);
  }
  const int rbase = r0 + ((lane >> 4) << 3);
#pragma unroll
  for (int r = 0; r < 8; ++r) {
    int row = rbase + r;
    if (row < Nr) out[(size_t)row * M + c0 + r16] = acc[r];
  }
}

// ---------------- attention logits: ALS/ALD[n,h] = <H[n,h,:], a_src/dst[h,:]> ----------------
__global__ void k_alpha(const float* __restrict__ H, const float* __restrict__ a_s,
                        const float* __restrict__ a_d, float* __restrict__ ALS,
                        float* __restrict__ ALD, int Nn, int HD, int C, int ld) {
  const int gw = (blockIdx.x * blockDim.x + threadIdx.x) >> 5;
  const int lane = threadIdx.x & 31;
  if (gw >= Nn * HD) return;
  const int n = gw / HD, h = gw - n * HD;
  const float* hp = H + (size_t)n * ld + h * C;
  const float* as = a_s + h * C;
  const float* ad = a_d + h * C;
  float ss = 0.f, sd = 0.f;
  for (int c = lane; c < C; c += 32) {
    float v = hp[c];
    ss += v * as[c];
    sd += v * ad[c];
  }
  for (int off = 16; off; off >>= 1) {
    ss += __shfl_xor(ss, off, 32);
    sd += __shfl_xor(sd, off, 32);
  }
  if (lane == 0) { ALS[gw] = ss; ALD[gw] = sd; }
}

// ---------------- edge passes ----------------
__device__ __forceinline__ void atomicMaxF(float* addr, float val) {
  if (val >= 0.f) atomicMax((int*)addr, __float_as_int(val));
  else            atomicMin((unsigned int*)addr, __float_as_uint(val));
}

__device__ __forceinline__ void edge_sd(const long long* ei, int e, int E, int* s, int* d) {
  if (e < E) { *s = (int)ei[e]; *d = (int)ei[E + e]; }
  else       { *s = *d = e - E; }  // appended self loops
}

__global__ void k_edge_max(const long long* __restrict__ ei, const float* __restrict__ ALS,
                           const float* __restrict__ ALD, float* __restrict__ Mx,
                           int E, int Etot, int HD) {
  int e = blockIdx.x * blockDim.x + threadIdx.x;
  if (e >= Etot) return;
  int s, d; edge_sd(ei, e, E, &s, &d);
  for (int h = 0; h < HD; ++h) {
    float v = ALS[s * HD + h] + ALD[d * HD + h];
    v = v > 0.f ? v : NEG_SLOPE * v;          // leaky_relu
    atomicMaxF(&Mx[d * HD + h], v);
  }
}

__global__ void k_edge_exp(const long long* __restrict__ ei, const float* __restrict__ ALS,
                           const float* __restrict__ ALD, const float* __restrict__ Mx,
                           float* __restrict__ Sx, float* __restrict__ EEXP,
                           int E, int Etot, int HD) {
  int e = blockIdx.x * blockDim.x + threadIdx.x;
  if (e >= Etot) return;
  int s, d; edge_sd(ei, e, E, &s, &d);
  for (int h = 0; h < HD; ++h) {
    float v = ALS[s * HD + h] + ALD[d * HD + h];
    v = v > 0.f ? v : NEG_SLOPE * v;
    float ex = expf(v - Mx[d * HD + h]);
    EEXP[(size_t)e * HD + h] = ex;
    atomicAdd(&Sx[d * HD + h], ex);
  }
}

// one 128-thread block per edge; thread c handles channel c of every head
__global__ void k_edge_aggr(const long long* __restrict__ ei, const float* __restrict__ H,
                            const float* __restrict__ EEXP, const float* __restrict__ Sx,
                            float* __restrict__ ACC, int E, int Etot, int HD, int C) {
  int e = blockIdx.x;
  if (e >= Etot) return;
  int s, d; edge_sd(ei, e, E, &s, &d);
  const int c = threadIdx.x;
  const int F = HD * C;
  const float* hs = H + (size_t)s * F;
  float* ao = ACC + (size_t)d * F;
  for (int h = 0; h < HD; ++h) {
    float alpha = EEXP[(size_t)e * HD + h] / Sx[d * HD + h];
    atomicAdd(&ao[h * C + c], hs[h * C + c] * alpha);
  }
}

// ---------------- finalize ----------------
__global__ void k_final_elu_bf16(const float* __restrict__ ACC, const float* __restrict__ b,
                                 __bf16* __restrict__ Anext, int n, int F) {
  int i = blockIdx.x * blockDim.x + threadIdx.x;
  if (i >= n) return;
  float v = ACC[i] + b[i % F];
  v = v > 0.f ? v : (expf(v) - 1.f);          // elu
  Anext[i] = (__bf16)v;
}

__global__ void k_final_out(const float* __restrict__ ACC, const float* __restrict__ b,
                            float* __restrict__ out, int n, int F) {
  int i = blockIdx.x * blockDim.x + threadIdx.x;
  if (i >= n) return;
  out[i] = ACC[i] + b[i % F];                 // heads=1, mean == identity
}

// ---------------- host ----------------
extern "C" void kernel_launch(void* const* d_in, const int* in_sizes, int n_in,
                              void* d_out, int out_size, void* d_ws, size_t ws_size,
                              hipStream_t stream) {
  const float*      x   = (const float*)d_in[0];
  const long long*  ei  = (const long long*)d_in[1];   // jnp.int64
  const float* W1 = (const float*)d_in[2];
  const float* a1s = (const float*)d_in[3];
  const float* a1d = (const float*)d_in[4];
  const float* b1  = (const float*)d_in[5];
  const float* W2 = (const float*)d_in[6];
  const float* a2s = (const float*)d_in[7];
  const float* a2d = (const float*)d_in[8];
  const float* b2  = (const float*)d_in[9];
  const float* W3 = (const float*)d_in[10];
  const float* a3s = (const float*)d_in[11];
  const float* a3d = (const float*)d_in[12];
  const float* b3  = (const float*)d_in[13];

  const int Nn   = in_sizes[0] / HIDC;   // 20000
  const int E    = in_sizes[1] / 2;      // 320000
  const int Etot = E + Nn;               // + self loops
  const int FMAX = 4 * HIDC;             // 512

  // workspace carve-up (256B aligned)
  char* ws = (char*)d_ws;
  size_t off = 0;
  auto carve = [&](size_t bytes) { char* p = ws + off; off = (off + bytes + 255) & ~(size_t)255; return p; };
  __bf16* Abf  = (__bf16*)carve((size_t)Nn * FMAX * sizeof(__bf16));
  __bf16* Wbf  = (__bf16*)carve((size_t)FMAX * FMAX * sizeof(__bf16));
  float*  Hb   = (float*) carve((size_t)Nn * FMAX * sizeof(float));
  float*  ACC  = (float*) carve((size_t)Nn * FMAX * sizeof(float));
  float*  ALS  = (float*) carve((size_t)Nn * 4 * sizeof(float));
  float*  ALD  = (float*) carve((size_t)Nn * 4 * sizeof(float));
  float*  Mx   = (float*) carve((size_t)Nn * 4 * sizeof(float));
  float*  Sx   = (float*) carve((size_t)Nn * 4 * sizeof(float));
  float*  EEXP = (float*) carve((size_t)Etot * 4 * sizeof(float));
  (void)ws_size; (void)n_in; (void)out_size;

  auto cdiv = [](long a, long b) { return (int)((a + b - 1) / b); };

  auto layer = [&](int K, int Mo, int HD, const float* W,
                   const float* a_s, const float* a_d) {
    const int F = HD * HIDC;             // == Mo
    k_cvt_bf16<<<cdiv((long)Mo * K, 256), 256, 0, stream>>>(W, Wbf, Mo * K);
    const int tiles = ((Nn + 15) / 16) * (Mo / 16);
    k_gemm_wmma<<<cdiv(tiles, 8), 256, 0, stream>>>(Abf, Wbf, Hb, Nn, K, Mo);
    k_alpha<<<cdiv((long)Nn * HD * 32, 256), 256, 0, stream>>>(Hb, a_s, a_d, ALS, ALD,
                                                               Nn, HD, HIDC, F);
    k_fill<<<cdiv((long)Nn * HD, 256), 256, 0, stream>>>(Mx, -INFINITY, Nn * HD);
    k_fill<<<cdiv((long)Nn * HD, 256), 256, 0, stream>>>(Sx, 0.f, Nn * HD);
    k_fill<<<cdiv((long)Nn * F, 256), 256, 0, stream>>>(ACC, 0.f, Nn * F);
    k_edge_max<<<cdiv(Etot, 256), 256, 0, stream>>>(ei, ALS, ALD, Mx, E, Etot, HD);
    k_edge_exp<<<cdiv(Etot, 256), 256, 0, stream>>>(ei, ALS, ALD, Mx, Sx, EEXP, E, Etot, HD);
    k_edge_aggr<<<Etot, HIDC, 0, stream>>>(ei, Hb, EEXP, Sx, ACC, E, Etot, HD, HIDC);
  };

  // layer 1: in=128, out=4*128 concat
  k_cvt_bf16<<<cdiv((long)Nn * HIDC, 256), 256, 0, stream>>>(x, Abf, Nn * HIDC);
  layer(HIDC, 4 * HIDC, 4, W1, a1s, a1d);
  k_final_elu_bf16<<<cdiv((long)Nn * FMAX, 256), 256, 0, stream>>>(ACC, b1, Abf, Nn * FMAX, FMAX);

  // layer 2: in=512, out=4*128 concat
  layer(4 * HIDC, 4 * HIDC, 4, W2, a2s, a2d);
  k_final_elu_bf16<<<cdiv((long)Nn * FMAX, 256), 256, 0, stream>>>(ACC, b2, Abf, Nn * FMAX, FMAX);

  // layer 3: in=512, out=128, heads=1, mean (== identity)
  layer(4 * HIDC, HIDC, 1, W3, a3s, a3d);
  k_final_out<<<cdiv((long)Nn * HIDC, 256), 256, 0, stream>>>(ACC, b3, (float*)d_out,
                                                              Nn * HIDC, HIDC);
}
